// LMU_24945170055971
// MI455X (gfx1250) — compile-verified, matrix operand order checked by reading, not run on hardware
//
#include <hip/hip_runtime.h>
#include <hip/hip_bf16.h>

// LMU (Legendre Memory Unit) on MI455X / gfx1250.
//
// Persistent kernel, 2 grid barriers per timestep.
//   Precompute (per launch):  M1 = (I + AT^T) @ W_m^T  (512x1024 f32),
//                             w_u = BT^T @ W_m^T       (1024 f32),
//                             bf16 WMMA-fragment-packed W_h^T, AT^T, M1.
//   Per step t:
//     phase A: u[b] = x*e_x + h.e_h + m.e_m        (128 wave reductions)
//     barrier
//     phase B: h' = tanh(x W_x^T + h@Wh^T + m@M1 + u*w_u)   [512 tiles]
//              m' = m + m@AT^T + u*BT                         [256 tiles]
//              (v_wmma_f32_16x16x32_bf16, f32 accumulate, f32 state)
//     barrier
//   Final: logits + softmax -> d_out (128x10 f32).
//
// The time loop is unrolled x2 with buffer parity specialized at the call
// site, so every state pointer traces directly to a kernarg pointer ->
// address-space inference gives global_load_b128 (NOT flat_load, which
// would also consume DScnt and the LDS path).

#define T_STEPS 784
#define BATCH   128
#define UNITS   1024
#define ORDER   512
#define NBLK    96          // persistent blocks (96*8 waves = 768 = tiles/step)
#define TPB     256

typedef __attribute__((ext_vector_type(16))) __bf16 v16bf;
typedef __attribute__((ext_vector_type(8)))  float  v8f;

union Frag16 { uint4 q[2]; v16bf v; };

__device__ __forceinline__ unsigned short f2bf(float f) {
  unsigned u = __builtin_bit_cast(unsigned, f);
  u += 0x7FFFu + ((u >> 16) & 1u);        // round-to-nearest-even
  return (unsigned short)(u >> 16);
}

// A fragment (16x32 bf16, M=rows) from row-major [rows][ld] bf16 buffer.
// Lane L holds row M = L&15; lanes<16 hold K {0..7,16..23}, lanes>=16 hold
// K {8..15,24..31} (CDNA5 16-bit A layout) -> two b128 loads per lane.
__device__ __forceinline__ v16bf load_a(const unsigned short* __restrict__ base,
                                        int row0, int k0, int ld, int lane) {
  const int r   = lane & 15;
  const int off = (lane < 16) ? 0 : 8;
  const unsigned short* p = base + (size_t)(row0 + r) * ld + k0 + off;
  Frag16 f;
  f.q[0] = *(const uint4*)(p);
  f.q[1] = *(const uint4*)(p + 16);
  return f.v;
}

// B fragment from fragment-packed weights: tile-major, 32 lanes x 16 bf16
// contiguous per lane (packed once at launch -> two fully-coalesced b128).
__device__ __forceinline__ v16bf load_b(const unsigned short* __restrict__ packed,
                                        int tile, int lane) {
  const unsigned short* p = packed + ((size_t)tile * 32 + lane) * 16;
  Frag16 f;
  f.q[0] = *(const uint4*)(p);
  f.q[1] = *(const uint4*)(p + 8);
  return f.v;
}

struct P {
  const float* __restrict__ x;
  const float* __restrict__ e_x;
  const float* __restrict__ e_h;
  const float* __restrict__ e_m;
  const float* __restrict__ W_x;
  const float* __restrict__ BT;
  const float* __restrict__ W_dense;
  const float* __restrict__ b_dense;
  float* __restrict__ hF0;  float* __restrict__ hF1;   // f32 state (dbl buf)
  float* __restrict__ mF0;  float* __restrict__ mF1;
  unsigned short* __restrict__ hB0; unsigned short* __restrict__ hB1; // bf16
  unsigned short* __restrict__ mB0; unsigned short* __restrict__ mB1;
  float* __restrict__ u;
  float* __restrict__ wu;
  unsigned short* __restrict__ WhP;   // fragment-packed bf16 weights
  unsigned short* __restrict__ M1P;
  unsigned short* __restrict__ ATP;
  unsigned* __restrict__ counter;
  float* __restrict__ out;
};

__device__ __forceinline__ void grid_barrier(unsigned* counter, unsigned target) {
  __threadfence();
  __syncthreads();
  if (threadIdx.x == 0) {
    __hip_atomic_fetch_add(counter, 1u, __ATOMIC_ACQ_REL, __HIP_MEMORY_SCOPE_AGENT);
    while (__hip_atomic_load(counter, __ATOMIC_ACQUIRE, __HIP_MEMORY_SCOPE_AGENT) < target)
      __builtin_amdgcn_s_sleep(2);
  }
  __syncthreads();
  __threadfence();
}

// ---- phase A: u[b] = x*e_x + h.e_h + m.e_m (one wave per batch row) ----
__device__ __forceinline__ void phaseA(const P& p, int t,
                                       const float* __restrict__ hF,
                                       const float* __restrict__ mF,
                                       int gw, int NW, int lane) {
  for (int b = gw; b < BATCH; b += NW) {
    const float* __restrict__ hr = hF + (size_t)b * UNITS;
    const float* __restrict__ mr = mF + (size_t)b * ORDER;
    float s = 0.f;
    for (int i = lane; i < UNITS; i += 32) s += hr[i] * p.e_h[i];
    for (int i = lane; i < ORDER; i += 32) s += mr[i] * p.e_m[i];
    for (int o = 16; o > 0; o >>= 1) s += __shfl_down(s, o, 32);
    if (lane == 0) p.u[b] = p.x[t * BATCH + b] * p.e_x[0] + s;
  }
}

// ---- phase B: 512 h'-tiles + 256 m'-tiles, one 16x16 tile per wave ----
__device__ __forceinline__ void phaseB(const P& p, int t,
                                       const unsigned short* __restrict__ hBc,
                                       const unsigned short* __restrict__ mBc,
                                       const float* __restrict__ mFc,
                                       float* __restrict__ hFn,
                                       float* __restrict__ mFn,
                                       unsigned short* __restrict__ hBn,
                                       unsigned short* __restrict__ mBn,
                                       int gw, int NW, int lane) {
  for (int job = gw; job < 768; job += NW) {
    if (job < 512) {                       // h' tile: (mt, nt)
      const int mt = job >> 6, nt = job & 63;
      const int nn = nt * 16 + (lane & 15);
      const int rbase = mt * 16 + ((lane >= 16) ? 8 : 0);

      // hoist epilogue operands above the WMMA loops (latency hidden)
      const float wx = p.W_x[nn], wuv = p.wu[nn];
      float xv[8], uv[8];
      #pragma unroll
      for (int v = 0; v < 8; ++v) xv[v] = p.x[t * BATCH + rbase + v];
      #pragma unroll
      for (int v = 0; v < 8; ++v) uv[v] = p.u[rbase + v];

      v8f acc = {};
      #pragma unroll 4
      for (int kt = 0; kt < UNITS / 32; ++kt) {      // h @ Wh^T
        v16bf a = load_a(hBc, mt * 16, kt * 32, UNITS, lane);
        v16bf b = load_b(p.WhP, kt * 64 + nt, lane);
        acc = __builtin_amdgcn_wmma_f32_16x16x32_bf16(
            false, a, false, b, (short)0, acc, false, false);
      }
      #pragma unroll 4
      for (int kt = 0; kt < ORDER / 32; ++kt) {      // m @ M1
        v16bf a = load_a(mBc, mt * 16, kt * 32, ORDER, lane);
        v16bf b = load_b(p.M1P, kt * 64 + nt, lane);
        acc = __builtin_amdgcn_wmma_f32_16x16x32_bf16(
            false, a, false, b, (short)0, acc, false, false);
      }

      float hv[8];
      #pragma unroll
      for (int v = 0; v < 8; ++v)
        hv[v] = tanhf(acc[v] + xv[v] * wx + uv[v] * wuv);
      #pragma unroll
      for (int v = 0; v < 8; ++v) {
        hFn[(size_t)(rbase + v) * UNITS + nn] = hv[v];
        hBn[(size_t)(rbase + v) * UNITS + nn] = f2bf(hv[v]);
      }
    } else {                               // m' tile: (mt, ct)
      const int j = job - 512;
      const int mt = j >> 5, ct = j & 31;
      const int kk = ct * 16 + (lane & 15);
      const int rbase = mt * 16 + ((lane >= 16) ? 8 : 0);

      const float bt = p.BT[kk];
      float uv[8], mo[8];
      #pragma unroll
      for (int v = 0; v < 8; ++v) uv[v] = p.u[rbase + v];
      #pragma unroll
      for (int v = 0; v < 8; ++v) mo[v] = mFc[(size_t)(rbase + v) * ORDER + kk];

      v8f acc = {};
      #pragma unroll 4
      for (int kt = 0; kt < ORDER / 32; ++kt) {      // m @ AT^T
        v16bf a = load_a(mBc, mt * 16, kt * 32, ORDER, lane);
        v16bf b = load_b(p.ATP, kt * 32 + ct, lane);
        acc = __builtin_amdgcn_wmma_f32_16x16x32_bf16(
            false, a, false, b, (short)0, acc, false, false);
      }

      float mv[8];
      #pragma unroll
      for (int v = 0; v < 8; ++v) mv[v] = acc[v] + mo[v] + uv[v] * bt;
      #pragma unroll
      for (int v = 0; v < 8; ++v) {
        mFn[(size_t)(rbase + v) * ORDER + kk] = mv[v];
        mBn[(size_t)(rbase + v) * ORDER + kk] = f2bf(mv[v]);
      }
    }
  }
}

__global__ void __launch_bounds__(TPB) lmu_persistent(P p) {
  const int lane = threadIdx.x & 31;
  const int gw   = blockIdx.x * (TPB >> 5) + (threadIdx.x >> 5);
  const int NW   = gridDim.x * (TPB >> 5);
  unsigned bar = 0;

  // ---- init: zero state buffer 0 (f32 + bf16) ----
  for (int i = blockIdx.x * blockDim.x + threadIdx.x; i < BATCH * UNITS;
       i += gridDim.x * blockDim.x) { p.hF0[i] = 0.f; p.hB0[i] = 0; }
  for (int i = blockIdx.x * blockDim.x + threadIdx.x; i < BATCH * ORDER;
       i += gridDim.x * blockDim.x) { p.mF0[i] = 0.f; p.mB0[i] = 0; }
  bar++; grid_barrier(p.counter, bar * gridDim.x);

  // time loop unrolled x2: buffer parity specialized at each call site so
  // every state pointer is a direct kernarg pointer (=> global, not flat).
  for (int t = 0; t < T_STEPS; t += 2) {
    phaseA(p, t, p.hF0, p.mF0, gw, NW, lane);
    bar++; grid_barrier(p.counter, bar * gridDim.x);
    phaseB(p, t, p.hB0, p.mB0, p.mF0, p.hF1, p.mF1, p.hB1, p.mB1, gw, NW, lane);
    bar++; grid_barrier(p.counter, bar * gridDim.x);

    phaseA(p, t + 1, p.hF1, p.mF1, gw, NW, lane);
    bar++; grid_barrier(p.counter, bar * gridDim.x);
    phaseB(p, t + 1, p.hB1, p.mB1, p.mF1, p.hF0, p.mF0, p.hB0, p.mB0, gw, NW, lane);
    bar++; grid_barrier(p.counter, bar * gridDim.x);
  }

  // ---- final: logits + softmax (T_STEPS even -> final h is in hF0) ----
  for (int b = gw; b < BATCH; b += NW) {
    const float* __restrict__ hr = p.hF0 + (size_t)b * UNITS;
    float logit[10];
    for (int j = 0; j < 10; ++j) {
      float s = 0.f;
      for (int i = lane; i < UNITS; i += 32) s += hr[i] * p.W_dense[j * UNITS + i];
      for (int o = 16; o > 0; o >>= 1) s += __shfl_down(s, o, 32);
      if (lane == 0) logit[j] = s + p.b_dense[j];
    }
    if (lane == 0) {
      float mx = logit[0];
      for (int j = 1; j < 10; ++j) mx = fmaxf(mx, logit[j]);
      float e[10], den = 0.f;
      for (int j = 0; j < 10; ++j) { e[j] = expf(logit[j] - mx); den += e[j]; }
      for (int j = 0; j < 10; ++j) p.out[b * 10 + j] = e[j] / den;
    }
  }
}

// ---------------- per-launch precompute kernels ----------------

__global__ void k_zero(unsigned* c) { if (threadIdx.x == 0) *c = 0u; }

// M1[k][n] = W_m[n][k] + sum_j AT[j][k] * W_m[n][j]   (512 x 1024 f32)
__global__ void k_M1(const float* __restrict__ AT, const float* __restrict__ W_m,
                     float* __restrict__ M1) {
  const int e = blockIdx.x * blockDim.x + threadIdx.x;
  if (e >= ORDER * UNITS) return;
  const int k = e >> 10, n = e & (UNITS - 1);
  float s = W_m[(size_t)n * ORDER + k];
  for (int j = 0; j < ORDER; ++j)
    s += AT[(size_t)j * ORDER + k] * W_m[(size_t)n * ORDER + j];
  M1[(size_t)k * UNITS + n] = s;
}

// w_u[n] = sum_j BT[j] * W_m[n][j]
__global__ void k_wu(const float* __restrict__ BT, const float* __restrict__ W_m,
                     float* __restrict__ wu) {
  const int n = blockIdx.x * blockDim.x + threadIdx.x;
  if (n >= UNITS) return;
  float s = 0.f;
  for (int j = 0; j < ORDER; ++j) s += BT[j] * W_m[(size_t)n * ORDER + j];
  wu[n] = s;
}

// Pack a logical B-matrix (K x N GEMM operand, element (n,k) at
// src[n*strideN + k*strideK]) into WMMA 32x16 bf16 fragments:
// dst[((kt*NT+nt)*32 + lane)*16 + i], CDNA5 16-bit fragment K pattern.
__global__ void k_pack(const float* __restrict__ src, unsigned short* __restrict__ dst,
                       int NT, int KT, long strideN, long strideK) {
  const int e = blockIdx.x * blockDim.x + threadIdx.x;
  if (e >= KT * NT * 512) return;
  const int i    = e & 15;
  const int lane = (e >> 4) & 31;
  const int tile = e >> 9;
  const int nt = tile % NT, kt = tile / NT;
  const int n = nt * 16 + (lane & 15);
  const int koff = (lane < 16) ? ((i < 8) ? i : (8 + i))
                               : ((i < 8) ? (8 + i) : (16 + i));
  const int k = kt * 32 + koff;
  dst[e] = f2bf(src[(long)n * strideN + (long)k * strideK]);
}

// ---------------- launch ----------------

extern "C" void kernel_launch(void* const* d_in, const int* in_sizes, int n_in,
                              void* d_out, int out_size, void* d_ws, size_t ws_size,
                              hipStream_t stream) {
  (void)in_sizes; (void)n_in; (void)out_size; (void)ws_size;
  const float* x       = (const float*)d_in[0];
  const float* e_x     = (const float*)d_in[1];
  const float* e_h     = (const float*)d_in[2];
  const float* e_m     = (const float*)d_in[3];
  const float* W_x     = (const float*)d_in[4];
  const float* W_h     = (const float*)d_in[5];
  const float* W_m     = (const float*)d_in[6];
  const float* AT      = (const float*)d_in[7];
  const float* BT      = (const float*)d_in[8];
  const float* W_dense = (const float*)d_in[9];
  const float* b_dense = (const float*)d_in[10];

  char* w = (char*)d_ws;
  auto take = [&](size_t b) { char* r = w; w += (b + 255) & ~(size_t)255; return r; };

  P p{};
  p.counter = (unsigned*)take(256);
  p.hF0 = (float*)take((size_t)BATCH * UNITS * 4);
  p.hF1 = (float*)take((size_t)BATCH * UNITS * 4);
  p.mF0 = (float*)take((size_t)BATCH * ORDER * 4);
  p.mF1 = (float*)take((size_t)BATCH * ORDER * 4);
  p.hB0 = (unsigned short*)take((size_t)BATCH * UNITS * 2);
  p.hB1 = (unsigned short*)take((size_t)BATCH * UNITS * 2);
  p.mB0 = (unsigned short*)take((size_t)BATCH * ORDER * 2);
  p.mB1 = (unsigned short*)take((size_t)BATCH * ORDER * 2);
  p.u   = (float*)take(BATCH * 4);
  p.wu  = (float*)take(UNITS * 4);
  float* M1 = (float*)take((size_t)ORDER * UNITS * 4);
  p.WhP = (unsigned short*)take((size_t)UNITS * UNITS * 2);
  p.M1P = (unsigned short*)take((size_t)ORDER * UNITS * 2);
  p.ATP = (unsigned short*)take((size_t)ORDER * ORDER * 2);

  p.x = x; p.e_x = e_x; p.e_h = e_h; p.e_m = e_m; p.W_x = W_x;
  p.BT = BT; p.W_dense = W_dense; p.b_dense = b_dense;
  p.out = (float*)d_out;

  k_zero<<<1, 64, 0, stream>>>(p.counter);
  k_M1<<<(ORDER * UNITS + 255) / 256, 256, 0, stream>>>(AT, W_m, M1);
  k_wu<<<(UNITS + 255) / 256, 256, 0, stream>>>(BT, W_m, p.wu);
  // W_h^T as B: element (n,k) = W_h[n*1024 + k]
  k_pack<<<(32 * 64 * 512) / 256, 256, 0, stream>>>(W_h, p.WhP, 64, 32, UNITS, 1);
  // AT^T as B: element (n,k) = AT[n*512 + k]
  k_pack<<<(16 * 32 * 512) / 256, 256, 0, stream>>>(AT, p.ATP, 32, 16, ORDER, 1);
  // M1 as B: element (n,k) = M1[k*1024 + n]
  k_pack<<<(16 * 64 * 512) / 256, 256, 0, stream>>>(M1, p.M1P, 64, 16, 1, UNITS);

  lmu_persistent<<<NBLK, TPB, 0, stream>>>(p);
}
